// TrajPreLocalAttnLongTraffic_63230508532523
// MI455X (gfx1250) — compile-verified, matrix operand order checked by reading, not run on hardware
//
#include <hip/hip_runtime.h>

#define SEQ   2048
#define HID   1024
#define G4H   4096
#define INP   576
#define LOCE  512
#define NLOC  32001
#define NEGV  (-1.0e11f)

typedef __bf16 bf16;
typedef __attribute__((ext_vector_type(16))) __bf16 v16bf;
typedef __attribute__((ext_vector_type(8)))  float  v8f;
typedef __attribute__((ext_vector_type(4)))  unsigned int u32x4;
typedef __attribute__((ext_vector_type(8)))  int i32x8;
typedef __attribute__((ext_vector_type(4)))  int i32x4;

// ---------------------------------------------------------------------------
// WMMA fragment helpers (CDNA5 wave32 layouts, ISA 7.12.2)
// A (16x32 bf16, MxK): lane L: m=L&15, half=L>>4; elems 0..7 -> K=half*8+e,
//                      elems 8..15 -> K=16+half*8+(e-8).
// B (32x16 bf16, KxN) loaded from a row-major "B-transposed" array R[n][k]
// (symmetric layout with n=L&15).
// C/D f32 16x16: lane L: n=L&15, VGPR r -> m = 8*(L>>4)+r.
// ---------------------------------------------------------------------------
__device__ __forceinline__ v16bf frag_ld_bf16(const bf16* base, int row0, int ldk, int k0) {
  const int lane = threadIdx.x & 31;
  const int r    = row0 + (lane & 15);
  const int half = lane >> 4;
  const bf16* p  = base + (size_t)r * ldk + k0 + half * 8;
  union { uint4 u; bf16 b[8]; } lo, hi;
  lo.u = *(const uint4*)(p);
  hi.u = *(const uint4*)(p + 16);
  v16bf f;
#pragma unroll
  for (int e = 0; e < 8; ++e) { f[e] = lo.b[e]; f[e + 8] = hi.b[e]; }
  return f;
}

// B fragment from an LDS-staged fp32 slab (one 16-row slab per block), cvt'd
// to bf16 in-register. p0 = &slab[lane&15][klocal + (lane>>4)*8].
__device__ __forceinline__ v16bf frag_ld_lds_f32(const float* p0) {
  v16bf f;
#pragma unroll
  for (int e = 0; e < 8; ++e) { f[e] = (bf16)p0[e]; f[e + 8] = (bf16)p0[16 + e]; }
  return f;
}

__device__ __forceinline__ v8f wmma_bf16(v16bf a, v16bf b, v8f c) {
  return __builtin_amdgcn_wmma_f32_16x16x32_bf16(false, a, false, b, (short)0, c,
                                                 false, false);
}

__device__ __forceinline__ void tile_st_f32(float* base, int m0, int n0, size_t ldn,
                                            const v8f& c) {
  const int lane = threadIdx.x & 31;
  const int n  = n0 + (lane & 15);
  const int mb = m0 + (lane >> 4) * 8;
#pragma unroll
  for (int r = 0; r < 8; ++r) base[(size_t)(mb + r) * ldn + n] = c[r];
}

// ---------------------------------------------------------------------------
// TDM: async DMA of a [rows=16] x [KSTAGE fp32] 2D tile of W_fc into LDS.
// D# per ISA 8.3-8.5. 2D tensor: groups 2/3 zero. tensor_dim1 = NLOC-n0 gives
// hardware zero-fill for the ragged last N tile. Tracked with TENSORcnt.
// ---------------------------------------------------------------------------
#define FC_KSTAGE 256

__device__ __forceinline__ void tdm_load_slab(const float* gptr, unsigned lds_off,
                                              int rows_valid) {
  unsigned long long ga = (unsigned long long)(size_t)gptr;
  u32x4 g0;
  g0[0] = 1u;                                           // count=1, user mode
  g0[1] = lds_off;                                      // LDS byte address
  g0[2] = (unsigned)(ga & 0xffffffffu);                 // global_addr[31:0]
  g0[3] = (unsigned)((ga >> 32) & 0x1ffffffu) | (2u << 30);  // addr[56:32]|type=2
  i32x8 g1;
  unsigned td0 = FC_KSTAGE;                             // tensor_dim0 (elems)
  unsigned td1 = (unsigned)rows_valid;                  // tensor_dim1 (rows)
  g1[0] = (int)(2u << 16);                              // data_size=2 (4 bytes)
  g1[1] = (int)((td0 & 0xffffu) << 16);                 // tensor_dim0[15:0]
  g1[2] = (int)((td0 >> 16) | ((td1 & 0xffffu) << 16)); // td0[31:16] | td1[15:0]
  g1[3] = (int)((td1 >> 16) | ((unsigned)FC_KSTAGE << 16)); // td1[31:16]|tile_dim0
  g1[4] = 16;                                           // tile_dim1=16, tile_dim2=0
  g1[5] = 2 * HID;                                      // tensor_dim0_stride[31:0]
  g1[6] = 0;                                            // stride hi | t1stride lo
  g1[7] = 0;
  i32x4 z4 = {0, 0, 0, 0};
#if __clang_major__ >= 23
  i32x8 z8 = {0, 0, 0, 0, 0, 0, 0, 0};
  __builtin_amdgcn_tensor_load_to_lds(g0, g1, z4, z4, z8, 0);
#else
  __builtin_amdgcn_tensor_load_to_lds(g0, g1, z4, z4, 0);
#endif
}

// ---------------------------------------------------------------------------
// small elementwise kernels
// ---------------------------------------------------------------------------
__global__ void k_cvt_bf16(const float* __restrict__ s, bf16* __restrict__ d, size_t n) {
  size_t i = (size_t)blockIdx.x * blockDim.x + threadIdx.x;
  if (i < n) d[i] = (bf16)s[i];
}

__global__ void k_bias(const float* bi, const float* bh, float* b) {
  int i = blockIdx.x * blockDim.x + threadIdx.x;
  if (i < G4H) b[i] = bi[i] + bh[i];
}

__global__ void k_embed(const int* __restrict__ loc, const int* __restrict__ tim,
                        const float* __restrict__ eloc, const float* __restrict__ etim,
                        bf16* __restrict__ x) {
  const int t = blockIdx.x;
  const int l = loc[t], m = tim[t];
  for (int i = threadIdx.x; i < INP; i += blockDim.x) {
    float v = (i < LOCE) ? eloc[(size_t)l * LOCE + i] : etim[(size_t)m * 64 + (i - LOCE)];
    x[(size_t)t * INP + i] = (bf16)v;
  }
}

__global__ void k_init(unsigned* bar) {
  if (threadIdx.x < 2 && blockIdx.x == 0) bar[threadIdx.x] = 0u;
}

// hs(f32) -> hs_bf16 and hsT_bf16
__global__ void k_hscvt(const float* __restrict__ hs, bf16* __restrict__ hb,
                        bf16* __restrict__ hTb) {
  size_t idx = (size_t)blockIdx.x * blockDim.x + threadIdx.x;
  if (idx >= (size_t)SEQ * HID) return;
  int i = (int)(idx / HID), j = (int)(idx % HID);
  bf16 v = (bf16)hs[idx];
  hb[idx] = v;
  hTb[(size_t)j * SEQ + i] = v;
}

// ---------------------------------------------------------------------------
// xg = x @ W_ih^T + (b_ih+b_hh)   [2048 x 4096], K=576, bf16 WMMA
// ---------------------------------------------------------------------------
__global__ void k_xg(const bf16* __restrict__ xb, const bf16* __restrict__ wih,
                     const float* __restrict__ bias, float* __restrict__ xg) {
  const int wave = (int)((blockIdx.x * blockDim.x + threadIdx.x) >> 5);
  const int tn = wave % (G4H / 16);
  const int tm = wave / (G4H / 16);
  v8f acc = {};
  for (int k0 = 0; k0 < INP; k0 += 32) {
    v16bf a = frag_ld_bf16(xb,  tm * 16, INP, k0);
    v16bf b = frag_ld_bf16(wih, tn * 16, INP, k0);
    acc = wmma_bf16(a, b, acc);
  }
  const float bv = bias[tn * 16 + (threadIdx.x & 15)];
#pragma unroll
  for (int r = 0; r < 8; ++r) acc[r] += bv;
  tile_st_f32(xg, tm * 16, tn * 16, G4H, acc);
}

// ---------------------------------------------------------------------------
// Persistent LSTM: 128 blocks x 512 threads, one grid barrier per step.
// Block g owns rows [g*32, g*32+32) of W_hh (L2-resident). h/c replicated in
// LDS per block; gate math recomputed by every block so only one barrier per
// step is needed. z double-buffered in global scratch.
// ---------------------------------------------------------------------------
#define LSTM_BLOCKS 128
#define LSTM_THREADS 512

__global__ void k_lstm(const float* __restrict__ xg, const float* __restrict__ whh,
                       float* __restrict__ zbuf, float* __restrict__ hs,
                       unsigned* __restrict__ bar) {
  __shared__ float sh[HID];
  __shared__ float sc[HID];
  const int t = threadIdx.x, g = blockIdx.x;
  for (int j = t; j < HID; j += LSTM_THREADS) { sh[j] = 0.f; sc[j] = 0.f; }
  __syncthreads();

  const int row  = g * 32 + (t >> 4);   // 0..4095
  const int part = t & 15;              // 16 parts x 64 elems of K
  const float* wrow = whh + (size_t)row * HID + part * 64;
  volatile unsigned* vgen = bar + 1;

  for (int step = 0; step < SEQ; ++step) {
    // --- partial GEMV: z[row] = xg[step,row] + W_hh[row,:] . h ---
    float sum = 0.f;
    const float* hp = sh + part * 64;
#pragma unroll 16
    for (int k = 0; k < 64; ++k) sum = fmaf(wrow[k], hp[k], sum);
#pragma unroll
    for (int off = 8; off; off >>= 1) sum += __shfl_xor(sum, off, 16);
    float* zb = zbuf + (size_t)(step & 1) * G4H;
    if (part == 0) zb[row] = sum + xg[(size_t)step * G4H + row];

    // --- grid barrier (sense via monotonically increasing generation) ---
    __syncthreads();
    if (t == 0) {
      unsigned my = *vgen;
      __threadfence();
      if (atomicAdd(bar, 1u) == LSTM_BLOCKS - 1) {
        atomicExch(bar, 0u);
        __threadfence();
        atomicAdd((unsigned*)(bar + 1), 1u);
      } else {
        while (*vgen == my) __builtin_amdgcn_s_sleep(2);
      }
    }
    __syncthreads();

    // --- gates (all blocks recompute full h into their own LDS) ---
    volatile const float* zv = zb;
    for (int j = t; j < HID; j += LSTM_THREADS) {
      float zi = zv[j], zf = zv[j + HID], zg = zv[j + 2 * HID], zo = zv[j + 3 * HID];
      float ii = 1.f / (1.f + __expf(-zi));
      float ff = 1.f / (1.f + __expf(-zf));
      float gg = tanhf(zg);
      float oo = 1.f / (1.f + __expf(-zo));
      float cn = ff * sc[j] + ii * gg;
      float hn = oo * tanhf(cn);
      sc[j] = cn; sh[j] = hn;
      if (g == 0) hs[(size_t)step * HID + j] = hn;
    }
    __syncthreads();
  }
}

// ---------------------------------------------------------------------------
// energies = hs @ hs^T, causal: strict-upper tiles filled with NEGV (no WMMA),
// diagonal tiles masked per-lane. K=1024.
// ---------------------------------------------------------------------------
__global__ void k_energy(const bf16* __restrict__ hb, float* __restrict__ E) {
  const int wave = (int)((blockIdx.x * blockDim.x + threadIdx.x) >> 5);
  const int nt = SEQ / 16;
  const int tj = wave % nt, ti = wave / nt;
  const int lane = threadIdx.x & 31;
  const int n = lane & 15, mb = (lane >> 4) * 8;
  if (tj > ti) {  // wave-uniform branch: EXEC stays all-ones on the WMMA path
#pragma unroll
    for (int r = 0; r < 8; ++r)
      E[(size_t)(ti * 16 + mb + r) * SEQ + tj * 16 + n] = NEGV;
    return;
  }
  v8f acc = {};
  for (int k0 = 0; k0 < HID; k0 += 32) {
    v16bf a = frag_ld_bf16(hb, ti * 16, HID, k0);
    v16bf b = frag_ld_bf16(hb, tj * 16, HID, k0);
    acc = wmma_bf16(a, b, acc);
  }
  if (ti == tj) {
#pragma unroll
    for (int r = 0; r < 8; ++r) if (n > mb + r) acc[r] = NEGV;
  }
  tile_st_f32(E, ti * 16, tj * 16, SEQ, acc);
}

// row softmax over [SEQ] -> bf16 attn
__global__ void k_softmax(const float* __restrict__ E, bf16* __restrict__ A) {
  __shared__ float red[8];
  const int row = blockIdx.x;
  const float* e = E + (size_t)row * SEQ;
  float mx = -3.4e38f;
  for (int j = threadIdx.x; j < SEQ; j += 256) mx = fmaxf(mx, e[j]);
#pragma unroll
  for (int o = 16; o; o >>= 1) mx = fmaxf(mx, __shfl_xor(mx, o, 32));
  if ((threadIdx.x & 31) == 0) red[threadIdx.x >> 5] = mx;
  __syncthreads();
  mx = red[0];
#pragma unroll
  for (int i = 1; i < 8; ++i) mx = fmaxf(mx, red[i]);
  __syncthreads();
  float s = 0.f;
  for (int j = threadIdx.x; j < SEQ; j += 256) s += __expf(e[j] - mx);
#pragma unroll
  for (int o = 16; o; o >>= 1) s += __shfl_xor(s, o, 32);
  if ((threadIdx.x & 31) == 0) red[threadIdx.x >> 5] = s;
  __syncthreads();
  s = red[0];
#pragma unroll
  for (int i = 1; i < 8; ++i) s += red[i];
  const float inv = 1.f / s;
  for (int j = threadIdx.x; j < SEQ; j += 256)
    A[(size_t)row * SEQ + j] = (bf16)(__expf(e[j] - mx) * inv);
}

// ---------------------------------------------------------------------------
// context = attn @ hs  [2048 x 1024]; B from hsT (row-major [HID][SEQ]).
// K-loop truncated at causal boundary (attn beyond it is exactly 0).
// ---------------------------------------------------------------------------
__global__ void k_context(const bf16* __restrict__ attn, const bf16* __restrict__ hT,
                          bf16* __restrict__ ctx) {
  const int wave = (int)((blockIdx.x * blockDim.x + threadIdx.x) >> 5);
  const int nt = HID / 16;
  const int tn = wave % nt, tm = wave / nt;
  const int kmax = min(SEQ, ((tm + 1) * 16 + 31) & ~31);
  v8f acc = {};
  for (int k0 = 0; k0 < kmax; k0 += 32) {
    v16bf a = frag_ld_bf16(attn, tm * 16, SEQ, k0);
    v16bf b = frag_ld_bf16(hT,   tn * 16, SEQ, k0);
    acc = wmma_bf16(a, b, acc);
  }
  const int lane = threadIdx.x & 31;
  const int n = tn * 16 + (lane & 15);
  const int mb = tm * 16 + (lane >> 4) * 8;
#pragma unroll
  for (int r = 0; r < 8; ++r) ctx[(size_t)(mb + r) * HID + n] = (bf16)acc[r];
}

// ---------------------------------------------------------------------------
// y = [hs|ctx] @ W_fc^T + b_fc  [2048 x 32001], K=2048.
// tn-major wave order: all 32 m-group waves for one W_fc slab are adjacent
// (4 blocks) -> slab read from HBM once (~262 MB total ~ 11 us @ 23.3 TB/s).
// All 8 waves of a block share tn, so the 16-row x 256-k fp32 W_fc slab is
// DMA'd into LDS once per block by the TDM (double-buffered, TENSORcnt),
// overlapping the next stage's DMA with this stage's 8x4 WMMAs. B fragments
// come from LDS with in-register fp32->bf16 cvt; ragged last tile zero-filled
// by TDM OOB semantics. M register-blocked x4: one B fragment feeds 4 WMMAs.
// ---------------------------------------------------------------------------
#define NTN 2001                    // ceil(32001/16)
#define FC_MB 4                     // M tiles per wave
#define FC_MG (SEQ / 16 / FC_MB)    // 32 M-groups
#define FC_NSTAGE (2 * HID / FC_KSTAGE)  // 8 stages of 256 k

__global__ void k_fc(const bf16* __restrict__ hb, const bf16* __restrict__ ctx,
                     const float* __restrict__ wfc, const float* __restrict__ bfc,
                     float* __restrict__ y) {
  __shared__ float sbuf[2][16 * FC_KSTAGE];   // 2 x 16 KB
  const int wave = (int)((blockIdx.x * blockDim.x + threadIdx.x) >> 5);
  const int tn   = wave / FC_MG;        // block-uniform (8 waves per block, 4 blocks/tn)
  const int tmg  = wave % FC_MG;
  const int m0   = tmg * FC_MB * 16;
  const int n0   = tn * 16;
  const int rows_valid = NLOC - n0;     // >=1; TDM zero-fills rows beyond this

  v8f acc[FC_MB];
#pragma unroll
  for (int i = 0; i < FC_MB; ++i) acc[i] = (v8f){};

  const float* gslab = wfc + (size_t)n0 * (2 * HID);
  if (threadIdx.x < 32)
    tdm_load_slab(gslab, (unsigned)(size_t)&sbuf[0][0], rows_valid);

  const int lane = threadIdx.x & 31;
  for (int s = 0; s < FC_NSTAGE; ++s) {
    if (threadIdx.x < 32) {
      if (s + 1 < FC_NSTAGE) {
        tdm_load_slab(gslab + (s + 1) * FC_KSTAGE,
                      (unsigned)(size_t)&sbuf[(s + 1) & 1][0], rows_valid);
        __builtin_amdgcn_s_wait_tensorcnt(1);   // current stage done, next in flight
      } else {
        __builtin_amdgcn_s_wait_tensorcnt(0);
      }
    }
    __syncthreads();                             // slab visible to all waves

    const float* brow = &sbuf[s & 1][0] + (lane & 15) * FC_KSTAGE + (lane >> 4) * 8;
#pragma unroll
    for (int kc = 0; kc < FC_KSTAGE / 32; ++kc) {
      const int kg = s * FC_KSTAGE + kc * 32;    // global k (stage-uniform half)
      v16bf b = frag_ld_lds_f32(brow + kc * 32);
      const bf16* abase = (kg < HID) ? hb : ctx;
      const int   ak    = (kg < HID) ? kg : kg - HID;
#pragma unroll
      for (int i = 0; i < FC_MB; ++i) {
        v16bf a = frag_ld_bf16(abase, m0 + i * 16, HID, ak);
        acc[i] = wmma_bf16(a, b, acc[i]);
      }
    }
    __syncthreads();                             // reads done before buffer reuse
  }

  const int n = n0 + (lane & 15);
  if (n < NLOC) {
    const float bv = bfc[n];
#pragma unroll
    for (int i = 0; i < FC_MB; ++i) {
      const int mb = m0 + i * 16 + (lane >> 4) * 8;
#pragma unroll
      for (int r = 0; r < 8; ++r)
        y[(size_t)(mb + r) * NLOC + n] = acc[i][r] + bv;
    }
  }
}

// in-place log_softmax over rows of [NLOC]
__global__ void k_logsoftmax(float* __restrict__ y) {
  __shared__ float red[8];
  float* p = y + (size_t)blockIdx.x * NLOC;
  float mx = -3.4e38f;
  for (int j = threadIdx.x; j < NLOC; j += 256) mx = fmaxf(mx, p[j]);
#pragma unroll
  for (int o = 16; o; o >>= 1) mx = fmaxf(mx, __shfl_xor(mx, o, 32));
  if ((threadIdx.x & 31) == 0) red[threadIdx.x >> 5] = mx;
  __syncthreads();
  mx = red[0];
#pragma unroll
  for (int i = 1; i < 8; ++i) mx = fmaxf(mx, red[i]);
  __syncthreads();
  float s = 0.f;
  for (int j = threadIdx.x; j < NLOC; j += 256) s += __expf(p[j] - mx);
#pragma unroll
  for (int o = 16; o; o >>= 1) s += __shfl_xor(s, o, 32);
  if ((threadIdx.x & 31) == 0) red[threadIdx.x >> 5] = s;
  __syncthreads();
  s = red[0];
#pragma unroll
  for (int i = 1; i < 8; ++i) s += red[i];
  const float lse = mx + logf(s);
  for (int j = threadIdx.x; j < NLOC; j += 256) p[j] -= lse;
}

// ---------------------------------------------------------------------------
extern "C" void kernel_launch(void* const* d_in, const int* in_sizes, int n_in,
                              void* d_out, int out_size, void* d_ws, size_t ws_size,
                              hipStream_t stream) {
  const int*   loc   = (const int*)d_in[0];
  const int*   tim   = (const int*)d_in[1];
  const float* eloc  = (const float*)d_in[2];
  const float* etim  = (const float*)d_in[3];
  const float* W_ih  = (const float*)d_in[4];
  const float* W_hh  = (const float*)d_in[5];
  const float* b_ih  = (const float*)d_in[6];
  const float* b_hh  = (const float*)d_in[7];
  const float* W_fc  = (const float*)d_in[8];
  const float* b_fc  = (const float*)d_in[9];
  float* out = (float*)d_out;

  char* w = (char*)d_ws;
  size_t o = 0;
  auto alloc = [&](size_t bytes) -> void* {
    void* p = w + o;
    o = (o + bytes + 255) & ~(size_t)255;
    return p;
  };
  bf16*     wih_b  = (bf16*)alloc((size_t)G4H * INP * 2);
  float*    bias   = (float*)alloc((size_t)G4H * 4);
  bf16*     x_b    = (bf16*)alloc((size_t)SEQ * INP * 2);
  float*    xg     = (float*)alloc((size_t)SEQ * G4H * 4);
  float*    zbuf   = (float*)alloc((size_t)2 * G4H * 4);
  unsigned* bar    = (unsigned*)alloc(256);
  float*    hs     = (float*)alloc((size_t)SEQ * HID * 4);
  bf16*     hs_b   = (bf16*)alloc((size_t)SEQ * HID * 2);
  bf16*     hsT_b  = (bf16*)alloc((size_t)SEQ * HID * 2);
  float*    energ  = (float*)alloc((size_t)SEQ * SEQ * 4);
  bf16*     attn_b = (bf16*)alloc((size_t)SEQ * SEQ * 2);
  bf16*     ctx_b  = (bf16*)alloc((size_t)SEQ * HID * 2);
  (void)ws_size; (void)n_in; (void)in_sizes; (void)out_size;

  // prep: convert W_ih, fold biases, gather embeddings, reset barrier
  {
    size_t n = (size_t)G4H * INP;
    k_cvt_bf16<<<(unsigned)((n + 255) / 256), 256, 0, stream>>>(W_ih, wih_b, n);
  }
  k_bias<<<G4H / 256, 256, 0, stream>>>(b_ih, b_hh, bias);
  k_embed<<<SEQ, 256, 0, stream>>>(loc, tim, eloc, etim, x_b);
  k_init<<<1, 64, 0, stream>>>(bar);

  // xg GEMM: (2048/16)*(4096/16) = 32768 waves -> 4096 blocks of 8 waves
  k_xg<<<4096, 256, 0, stream>>>(x_b, wih_b, bias, xg);

  // persistent LSTM
  k_lstm<<<LSTM_BLOCKS, LSTM_THREADS, 0, stream>>>(xg, W_hh, zbuf, hs, bar);

  // hs -> bf16 (+ transpose copy)
  k_hscvt<<<(SEQ * HID) / 256, 256, 0, stream>>>(hs, hs_b, hsT_b);

  // energies: 128*128 tiles -> 2048 blocks
  k_energy<<<2048, 256, 0, stream>>>(hs_b, energ);
  k_softmax<<<SEQ, 256, 0, stream>>>(energ, attn_b);

  // context: 128*64 tiles -> 1024 blocks
  k_context<<<1024, 256, 0, stream>>>(attn_b, hsT_b, ctx_b);

  // final GEMM: 2001 N-tiles * 32 M-groups = 64032 waves -> 8004 blocks
  k_fc<<<(NTN * FC_MG) / 8, 256, 0, stream>>>(hs_b, ctx_b, W_fc, b_fc, out);
  k_logsoftmax<<<SEQ, 256, 0, stream>>>(out);
}
